// JointNet_2757369004490
// MI455X (gfx1250) — matrix operand order repaired and checked by reading, not compiled, not run on hardware
//
#include <hip/hip_runtime.h>

typedef __attribute__((ext_vector_type(16))) _Float16 v16h;
typedef __attribute__((ext_vector_type(8)))  _Float16 v8h;
typedef __attribute__((ext_vector_type(8)))  float    v8f;

#define B_   4
#define T_   256
#define U_   64
#define D_   640     // D_ENC == D_DEC == INNER
#define V_   1024    // VOCAB
#define KD   640     // K of the big GEMM
#define LSTR 648     // padded LDS row stride in halves (multiple of 8 -> 16B runs aligned)

// ---------------------------------------------------------------------------
// Kernel 1: exact fp32 projections (feed tanh, must stay full precision)
//   enc_proj[1024][640] = enc[1024][640] @ w1[0:640][:]
//   dec_proj[ 256][640] = dec[ 256][640] @ w1[640:1280][:]
// ---------------------------------------------------------------------------
__global__ void proj_kernel(const float* __restrict__ enc,
                            const float* __restrict__ dec,
                            const float* __restrict__ w1,
                            float* __restrict__ enc_proj,
                            float* __restrict__ dec_proj) {
  int g = blockIdx.x * blockDim.x + threadIdx.x;
  if (g >= (B_ * T_ + B_ * U_) * D_) return;
  int row = g / D_;
  int h   = g % D_;
  const float* x;
  const float* wb;
  float* o;
  if (row < B_ * T_) {
    x  = enc + row * D_;
    wb = w1;                      // rows 0..639
    o  = enc_proj + row * D_ + h;
  } else {
    int r2 = row - B_ * T_;
    x  = dec + r2 * D_;
    wb = w1 + D_ * D_;            // rows 640..1279
    o  = dec_proj + r2 * D_ + h;
  }
  float acc = 0.f;
#pragma unroll 4
  for (int i = 0; i < D_; ++i)
    acc = fmaf(x[i], wb[i * D_ + h], acc);
  *o = acc;
}

// ---------------------------------------------------------------------------
// Kernel 2: transpose w2 [640,1024] -> K-major [1024][640] and split into
// hi/lo f16 planes (hi = f16(v), lo = f16(v - hi)); hi+lo carries ~22 bits.
// ---------------------------------------------------------------------------
__global__ void w2split_kernel(const float* __restrict__ w2,
                               _Float16* __restrict__ w2t_hi,
                               _Float16* __restrict__ w2t_lo) {
  int g = blockIdx.x * blockDim.x + threadIdx.x;
  if (g >= V_ * KD) return;
  int n = g / KD;
  int k = g % KD;
  float v = w2[k * V_ + n];
  _Float16 hi = (_Float16)v;
  _Float16 lo = (_Float16)(v - (float)hi);
  w2t_hi[n * KD + k] = hi;
  w2t_lo[n * KD + k] = lo;
}

// ---------------------------------------------------------------------------
// Kernel 3: joint GEMM with on-the-fly tanh tile.
// One block (256 thr = 8 waves) per 16-row strip of the flattened [B*T*U] M
// dimension. Phase 1: h tile (16 x 640) computed once, tanh'd, and stored in
// LDS pre-split into hi/lo f16 planes. Phase 2: each wave owns 128 VOCAB
// columns; pure ds-load A fragments + global B fragments -> 3 WMMAs per
// (ntile, k-step), fp32 accumulators. No conversions in the hot loop.
// ---------------------------------------------------------------------------
__global__ __launch_bounds__(256, 1)
void joint_kernel(const float* __restrict__ enc_proj,
                  const float* __restrict__ dec_proj,
                  const float* __restrict__ b1,
                  const _Float16* __restrict__ w2t_hi,
                  const _Float16* __restrict__ w2t_lo,
                  const float* __restrict__ b2,
                  float* __restrict__ out) {
  __shared__ _Float16 s_hi[16 * LSTR];       // 20,736 B
  __shared__ _Float16 s_lo[16 * LSTR];       // 20,736 B

  const int bid   = blockIdx.x;              // 0..4095
  const int tid   = threadIdx.x;
  const int R0    = bid << 4;                // global M row base
  const int e_row = bid >> 2;                // (b,t) index: R0 / U
  const int bb    = bid >> 10;               // batch
  const int u0    = (bid & 3) << 4;
  const int dbase = bb * U_ + u0;            // dec row base

  // ---- Phase 1: h[r][k] = tanh(enc_proj + dec_proj + b1), split hi/lo ----
  const float* ep = enc_proj + e_row * D_;
  for (int k = tid; k < D_; k += 256) {
    const float e = ep[k] + b1[k];
#pragma unroll
    for (int r = 0; r < 16; ++r) {
      const float v = tanhf(e + dec_proj[(dbase + r) * D_ + k]);
      const _Float16 hi = (_Float16)v;
      s_hi[r * LSTR + k] = hi;
      s_lo[r * LSTR + k] = (_Float16)(v - (float)hi);
    }
  }
  __syncthreads();

  // ---- Phase 2: WMMA GEMM ----
  const int wave  = tid >> 5;
  const int lane  = tid & 31;
  const int lhalf = lane >> 4;               // 0 | 1
  const int l15   = lane & 15;

  for (int chunk = 0; chunk < 2; ++chunk) {
    const int n0 = wave * 128 + chunk * 64;
    v8f acc[4] = {};

#pragma unroll 2
    for (int ks = 0; ks < KD; ks += 32) {
      // A fragment (16x32 f16): lane<16 -> K {ks..+15},
      //                         lane>=16 -> K {ks+16..+31}
      const int kb = ks + (lhalf << 4);
      const _Float16* pa = &s_hi[l15 * LSTR + kb];
      const _Float16* pl = &s_lo[l15 * LSTR + kb];
      const v8h ah0 = *(const v8h*)pa;             // 16B ds load
      const v8h ah1 = *(const v8h*)(pa + 8);
      const v8h al0 = *(const v8h*)pl;
      const v8h al1 = *(const v8h*)(pl + 8);
      const v16h a_hi = __builtin_shufflevector(ah0, ah1,
          0, 1, 2, 3, 4, 5, 6, 7, 8, 9, 10, 11, 12, 13, 14, 15);
      const v16h a_lo = __builtin_shufflevector(al0, al1,
          0, 1, 2, 3, 4, 5, 6, 7, 8, 9, 10, 11, 12, 13, 14, 15);

      // B fragments: lanes 0-15 col N, K=ks..ks+15; lanes 16-31 K=ks+16..+31
      const int koff = ks + (lhalf << 4);
#pragma unroll
      for (int nt = 0; nt < 4; ++nt) {
        const int col = n0 + nt * 16 + l15;
        const v16h b_hi = *(const v16h*)(w2t_hi + col * KD + koff);  // 32B aligned
        const v16h b_lo = *(const v16h*)(w2t_lo + col * KD + koff);
        acc[nt] = __builtin_amdgcn_wmma_f32_16x16x32_f16(
            false, a_hi, false, b_hi, (short)0, acc[nt], false, false);
        acc[nt] = __builtin_amdgcn_wmma_f32_16x16x32_f16(
            false, a_hi, false, b_lo, (short)0, acc[nt], false, false);
        acc[nt] = __builtin_amdgcn_wmma_f32_16x16x32_f16(
            false, a_lo, false, b_hi, (short)0, acc[nt], false, false);
      }
    }

    // ---- Store: D layout: N = lane&15, M = j + 8*(lane>=16) ----
    const int mbase = R0 + (lhalf << 3);
#pragma unroll
    for (int nt = 0; nt < 4; ++nt) {
      const int ncol  = n0 + nt * 16 + l15;
      const float bia = b2[ncol];
#pragma unroll
      for (int j = 0; j < 8; ++j) {
        out[(size_t)(mbase + j) * V_ + ncol] = acc[nt][j] + bia;
      }
    }
  }
}

// ---------------------------------------------------------------------------
extern "C" void kernel_launch(void* const* d_in, const int* in_sizes, int n_in,
                              void* d_out, int out_size, void* d_ws, size_t ws_size,
                              hipStream_t stream) {
  (void)in_sizes; (void)n_in; (void)out_size; (void)ws_size;
  const float* enc = (const float*)d_in[0];  // [4,256,640]
  const float* dec = (const float*)d_in[1];  // [4,64,640]
  const float* w1  = (const float*)d_in[2];  // [1280,640]
  const float* b1  = (const float*)d_in[3];  // [640]
  const float* w2  = (const float*)d_in[4];  // [640,1024]
  const float* b2  = (const float*)d_in[5];  // [1024]
  float* out = (float*)d_out;                // [4,256,64,1024]

  char* ws = (char*)d_ws;
  float*    enc_proj = (float*)ws;                                   // 1024*640 f32
  float*    dec_proj = (float*)(ws + (size_t)1024 * 640 * 4);        //  256*640 f32
  _Float16* w2t_hi   = (_Float16*)(ws + (size_t)(1024 + 256) * 640 * 4);
  _Float16* w2t_lo   = w2t_hi + (size_t)1024 * 640;                  // total ~5.6 MB

  proj_kernel<<<(1280 * 640) / 256, 256, 0, stream>>>(enc, dec, w1, enc_proj, dec_proj);
  w2split_kernel<<<(1024 * 640) / 256, 256, 0, stream>>>(w2, w2t_hi, w2t_lo);
  joint_kernel<<<(B_ * T_ * U_) / 16, 256, 0, stream>>>(enc_proj, dec_proj, b1,
                                                        w2t_hi, w2t_lo, b2, out);
}
